// QuantPatchEmbedCNN_79182017069514
// MI455X (gfx1250) — 8-bit matrix operand order repaired and checked by reading, compiled, not run on hardware
//
#include <hip/hip_runtime.h>
#include <math.h>
#include <stdint.h>

typedef int   v8i __attribute__((ext_vector_type(8)));
typedef float v8f __attribute__((ext_vector_type(8)));
typedef float v2f __attribute__((ext_vector_type(2)));

constexpr int BATCH   = 512;
constexpr int CH      = 16;
constexpr int HH      = 704;
constexpr int WIN     = 40;
constexpr int WOUT    = 4;
constexpr int NPOS    = HH * WOUT;
constexpr int PER_B   = CH * NPOS;
constexpr int X_PER_B = HH * WIN;
constexpr int GROUPS  = 4;
constexpr int GSIZE   = PER_B / GROUPS;
constexpr long long TOTAL = (long long)BATCH * PER_B;

__device__ __forceinline__ void async_ld_lds_b128(uint32_t lds_off, const void* gaddr) {
  asm volatile("global_load_async_to_lds_b128 %0, %1, off"
               :: "v"(lds_off), "v"((uint64_t)(uintptr_t)gaddr)
               : "memory");
}
__device__ __forceinline__ void wait_async0() {
  asm volatile("s_wait_asynccnt 0" ::: "memory");
}

__device__ __forceinline__ float wave_sum(float v) {
#pragma unroll
  for (int o = 16; o > 0; o >>= 1) v += __shfl_down(v, o, 32);
  return v;
}
__device__ __forceinline__ float wave_max(float v) {
#pragma unroll
  for (int o = 16; o > 0; o >>= 1) v = fmaxf(v, __shfl_down(v, o, 32));
  return v;
}
__device__ __forceinline__ float block_sum(float v, float* red) {
  const int lane = threadIdx.x & 31, wid = threadIdx.x >> 5;
  v = wave_sum(v);
  if (lane == 0) red[wid] = v;
  __syncthreads();
  if (wid == 0) {
    float t = (lane < 8) ? red[lane] : 0.f;
    t = wave_sum(t);
    if (lane == 0) red[0] = t;
  }
  __syncthreads();
  float r = red[0];
  __syncthreads();
  return r;
}
__device__ __forceinline__ float block_max(float v, float* red) {
  const int lane = threadIdx.x & 31, wid = threadIdx.x >> 5;
  v = wave_max(v);
  if (lane == 0) red[wid] = v;
  __syncthreads();
  if (wid == 0) {
    float t = (lane < 8) ? red[lane] : 0.f;
    t = wave_max(t);
    if (lane == 0) red[0] = t;
  }
  __syncthreads();
  float r = red[0];
  __syncthreads();
  return r;
}

__device__ void gn_gelu_store(const float* hsv, float* red,
                              const float* __restrict__ gamma,
                              const float* __restrict__ beta,
                              float* __restrict__ outb,
                              unsigned int* __restrict__ max_slot) {
  const int tid = threadIdx.x;
  for (int g = 0; g < GROUPS; ++g) {
    float s = 0.f, ss = 0.f;
    const float* hg = hsv + g * GSIZE;
    for (int i = tid; i < GSIZE; i += 256) {
      float v = hg[i];
      s += v; ss += v * v;
    }
    s  = block_sum(s, red);
    ss = block_sum(ss, red);
    if (tid == 0) {
      float m   = s * (1.0f / (float)GSIZE);
      float var = ss * (1.0f / (float)GSIZE) - m * m;
      red[32 + g] = m;
      red[36 + g] = rsqrtf(var + 1e-5f);
    }
    __syncthreads();
  }
  float mx = 0.f;
  for (int i = tid; i < PER_B; i += 256) {
    int c = i / NPOS;
    int g = c >> 2;
    float v = (hsv[i] - red[32 + g]) * red[36 + g] * gamma[c] + beta[c];
    float y = 0.5f * v * (1.0f + erff(v * 0.70710678118654752f));
    outb[i] = y;
    mx = fmaxf(mx, fabsf(y));
  }
  mx = block_max(mx, red);
  if (tid == 0) atomicMax(max_slot, __float_as_uint(mx));
}

__global__ void __launch_bounds__(256)
k_weights(const float* __restrict__ w1, const float* __restrict__ w2,
          const float* __restrict__ w3, unsigned int* __restrict__ slots,
          float* __restrict__ sw, float* __restrict__ w1q,
          uint32_t* __restrict__ w2frag, uint32_t* __restrict__ w3frag) {
  __shared__ float red[64];
  const int tid = threadIdx.x;
  if (tid < 8) slots[tid] = 0u;

  float m = 0.f;
  for (int i = tid; i < CH * 19; i += 256) m = fmaxf(m, fabsf(w1[i]));
  m = block_max(m, red);
  float s1 = fmaxf(m, 1e-8f) * (1.0f / 127.0f);
  for (int i = tid; i < CH * 19; i += 256) {
    float q = fminf(fmaxf(rintf(w1[i] / s1), -127.f), 127.f);
    w1q[i] = q * s1;
  }

  const float* wsrc_[2] = {w2, w3};
  uint32_t* frag_[2]    = {w2frag, w3frag};
  for (int sel = 0; sel < 2; ++sel) {
    const float* wsrc = wsrc_[sel];
    m = 0.f;
    for (int i = tid; i < 768; i += 256) m = fmaxf(m, fabsf(wsrc[i]));
    m = block_max(m, red);
    float s = fmaxf(m, 1e-8f) * (1.0f / 127.0f);
    if (tid == 0) sw[sel] = s;
    const int lane = tid & 31, v = tid >> 5;
    const int half = lane >> 4, M = lane & 15;
    uint32_t d = 0;
#pragma unroll
    for (int bi = 0; bi < 4; ++bi) {
      int K = ((v >> 1) << 4) + (half << 3) + ((v & 1) << 2) + bi;
      int q = 0;
      if (K < 48) {
        float val = wsrc[M * 48 + K];
        q = (int)fminf(fmaxf(rintf(val / s), -127.f), 127.f);
      }
      d |= ((uint32_t)(q & 0xFF)) << (8 * bi);
    }
    frag_[sel][tid] = d;
  }
}

__global__ void __launch_bounds__(256)
k_conv1_gn(const float* __restrict__ x, const float* __restrict__ w1q_hdr,
           const float* __restrict__ gamma, const float* __restrict__ beta,
           float* __restrict__ out, unsigned int* __restrict__ max_slot) {
  extern __shared__ float smem[];
  float* xs  = smem;
  float* hsv = xs + X_PER_B;
  float* wq  = hsv + PER_B;
  float* red = wq + 304;
  const int tid  = threadIdx.x;
  const int b    = blockIdx.x;
  const int lane = tid & 31, wid = tid >> 5;

  const char* xsrc = (const char*)(x + (size_t)b * X_PER_B);
  const uint32_t xs_lds = (uint32_t)(uintptr_t)xs;
  for (int i = tid; i < X_PER_B / 4; i += 256)
    async_ld_lds_b128(xs_lds + i * 16, xsrc + i * 16);
  for (int i = tid; i < CH * 19; i += 256) wq[i] = w1q_hdr[i];
  wait_async0();
  __syncthreads();

  const int half = lane >> 4, n = lane & 15;
  v2f afr[5];
#pragma unroll
  for (int kk = 0; kk < 5; ++kk) {
    int k0 = kk * 4 + 2 * half;
    afr[kk][0] = (k0     < 19) ? wq[(lane & 15) * 19 + k0]     : 0.f;
    afr[kk][1] = (k0 + 1 < 19) ? wq[(lane & 15) * 19 + k0 + 1] : 0.f;
  }

  for (int t = wid; t < NPOS / 16; t += 8) {
    int pos = t * 16 + n;
    int h = pos >> 2, w = pos & 3;
    const float* xr = xs + h * WIN;
    int base = w * 10 - 9;
    v8f acc = {0.f, 0.f, 0.f, 0.f, 0.f, 0.f, 0.f, 0.f};
#pragma unroll
    for (int kk = 0; kk < 5; ++kk) {
      int k0  = kk * 4 + 2 * half;
      int xi0 = base + k0;
      v2f bf;
      bf[0] = (xi0     >= 0 && xi0     < WIN && k0     < 19) ? xr[xi0]     : 0.f;
      bf[1] = (xi0 + 1 >= 0 && xi0 + 1 < WIN && k0 + 1 < 19) ? xr[xi0 + 1] : 0.f;
      acc = __builtin_amdgcn_wmma_f32_16x16x4_f32(false, afr[kk], false, bf,
                                                  (short)0, acc, false, false);
    }
#pragma unroll
    for (int r = 0; r < 8; ++r)
      hsv[(r + (half << 3)) * NPOS + pos] = acc[r];
  }
  __syncthreads();
  gn_gelu_store(hsv, red, gamma, beta, out + (size_t)b * PER_B, max_slot);
}

__global__ void __launch_bounds__(256)
k_conv_wmma_gn(const int8_t* __restrict__ inq, const uint32_t* __restrict__ afrag,
               const float* __restrict__ sw_ptr, const unsigned int* __restrict__ in_max,
               const float* __restrict__ gamma, const float* __restrict__ beta,
               float* __restrict__ out, unsigned int* __restrict__ max_slot) {
  extern __shared__ float smem[];
  float* hsv = smem;
  float* red = hsv + PER_B;
  uint8_t* aq = (uint8_t*)(red + 64);
  const int tid  = threadIdx.x;
  const int b    = blockIdx.x;
  const int lane = tid & 31, wid = tid >> 5;

  const char* qsrc = (const char*)(inq + (size_t)b * PER_B);
  const uint32_t aq_lds = (uint32_t)(uintptr_t)aq;
  for (int i = tid; i < PER_B / 16; i += 256)
    async_ld_lds_b128(aq_lds + i * 16, qsrc + i * 16);

  v8i a;
#pragma unroll
  for (int v = 0; v < 8; ++v) a[v] = (int)afrag[v * 32 + lane];

  const float s_in  = fmaxf(__uint_as_float(*in_max), 1e-8f) * (1.0f / 127.0f);
  const float scale = s_in * (*sw_ptr);
  wait_async0();
  __syncthreads();

  const int half = lane >> 4, n = lane & 15;
  for (int t = wid; t < NPOS / 16; t += 8) {
    int pos = t * 16 + n;
    int h = pos >> 2, w = pos & 3;
    v8i bf;
#pragma unroll
    for (int v = 0; v < 8; ++v) {
      uint32_t d = 0;
#pragma unroll
      for (int bi = 0; bi < 4; ++bi) {
        int K = ((v >> 1) << 4) + (half << 3) + ((v & 1) << 2) + bi;
        uint32_t bv = 0;
        if (K < 48) {
          int ic = K / 3, j = K - 3 * ic;
          int wc = w + j - 1;
          if (wc >= 0 && wc < WOUT) bv = aq[ic * NPOS + h * WOUT + wc];
        }
        d |= bv << (8 * bi);
      }
      bf[v] = (int)d;
    }
    v8i acc = {0, 0, 0, 0, 0, 0, 0, 0};
    acc = __builtin_amdgcn_wmma_i32_16x16x64_iu8(true, a, true, bf, acc, false, false);
#pragma unroll
    for (int r = 0; r < 8; ++r) {
      int c = r + (half << 3);
      hsv[c * NPOS + pos] = (float)acc[r] * scale;
    }
  }
  __syncthreads();
  gn_gelu_store(hsv, red, gamma, beta, out + (size_t)b * PER_B, max_slot);
}

__global__ void __launch_bounds__(256)
k_quant_act(const float* __restrict__ in, const unsigned int* __restrict__ max_slot,
            int8_t* __restrict__ out, long long n) {
  float s   = fmaxf(__uint_as_float(*max_slot), 1e-8f) * (1.0f / 127.0f);
  float inv = 1.0f / s;
  long long stride = (long long)gridDim.x * blockDim.x;
  for (long long i = (long long)blockIdx.x * blockDim.x + threadIdx.x; i < n; i += stride) {
    __builtin_prefetch(in + i + stride, 0, 0);
    float q = fminf(fmaxf(rintf(in[i] * inv), -128.f), 127.f);
    out[i] = (int8_t)q;
  }
}
__global__ void __launch_bounds__(256)
k_quant_final(const float* __restrict__ in, const unsigned int* __restrict__ max_slot,
              float* __restrict__ out, long long n) {
  float s   = fmaxf(__uint_as_float(*max_slot), 1e-8f) * (1.0f / 127.0f);
  float inv = 1.0f / s;
  long long stride = (long long)gridDim.x * blockDim.x;
  for (long long i = (long long)blockIdx.x * blockDim.x + threadIdx.x; i < n; i += stride) {
    float q = fminf(fmaxf(rintf(in[i] * inv), -128.f), 127.f);
    out[i] = q * s;
  }
}

extern "C" void kernel_launch(void* const* d_in, const int* in_sizes, int n_in,
                              void* d_out, int out_size, void* d_ws, size_t ws_size,
                              hipStream_t stream) {
  (void)in_sizes; (void)n_in; (void)out_size; (void)ws_size;
  const float* x  = (const float*)d_in[0];
  const float* w1 = (const float*)d_in[1];
  const float* w2 = (const float*)d_in[2];
  const float* w3 = (const float*)d_in[3];
  const float* g1 = (const float*)d_in[4];
  const float* b1 = (const float*)d_in[5];
  const float* g2 = (const float*)d_in[6];
  const float* b2 = (const float*)d_in[7];
  const float* g3 = (const float*)d_in[8];
  const float* b3 = (const float*)d_in[9];

  char* ws = (char*)d_ws;
  unsigned int* slots = (unsigned int*)ws;
  float*    sw   = (float*)(ws + 64);
  float*    w1q  = (float*)(ws + 128);
  uint32_t* w2f  = (uint32_t*)(ws + 2048);
  uint32_t* w3f  = (uint32_t*)(ws + 3072);
  float*    hbuf = (float*)(ws + 8192);
  int8_t*   qbuf = (int8_t*)(ws + 8192 + (size_t)TOTAL * 4);

  const size_t smem1 = (size_t)(X_PER_B + PER_B + 304 + 64) * 4;
  const size_t smem2 = (size_t)(PER_B + 64) * 4 + PER_B;

  k_weights<<<1, 256, 0, stream>>>(w1, w2, w3, slots, sw, w1q, w2f, w3f);
  k_conv1_gn<<<BATCH, 256, smem1, stream>>>(x, w1q, g1, b1, hbuf, slots + 0);
  k_quant_act<<<2048, 256, 0, stream>>>(hbuf, slots + 0, qbuf, TOTAL);
  k_conv_wmma_gn<<<BATCH, 256, smem2, stream>>>(qbuf, w2f, sw + 0, slots + 0,
                                                g2, b2, hbuf, slots + 1);
  k_quant_act<<<2048, 256, 0, stream>>>(hbuf, slots + 1, qbuf, TOTAL);
  k_conv_wmma_gn<<<BATCH, 256, smem2, stream>>>(qbuf, w3f, sw + 1, slots + 1,
                                                g3, b3, hbuf, slots + 2);
  k_quant_final<<<2048, 256, 0, stream>>>(hbuf, slots + 2, (float*)d_out, TOTAL);
}
